// Rotate_Conv_35845797052770
// MI455X (gfx1250) — matrix operand order repaired and checked by reading, not compiled, not run on hardware
//
#include <hip/hip_runtime.h>
#include <hip/hip_bf16.h>

typedef __attribute__((ext_vector_type(16))) _Float16 v16h;
typedef __attribute__((ext_vector_type(8)))  float    v8f;
typedef __attribute__((ext_vector_type(4)))  unsigned int v4u;

union F16Frag { v16h h; v4u u[2]; };

#define CIN    64
#define COUT   256
#define HH     96
#define WW     96
#define KDIM   576          // 64 * 9, == 18 * 32
#define ROWLEN 72           // padded LDS row length (halves); multiple of 8 keeps 16B alignment
#define LDS_ELEMS (98 * 3 * ROWLEN)

// ---------------------------------------------------------------------------
// Phase A: build rotated conv weights, repack to f16 A[oc][k], k = tap*64 + c.
// Also precompute BN scale/shift.
// ---------------------------------------------------------------------------
__global__ void rc_build_kernel(const float* __restrict__ w,
                                const float* __restrict__ gamma,
                                const float* __restrict__ beta,
                                const float* __restrict__ rmean,
                                const float* __restrict__ rvar,
                                _Float16* __restrict__ A,
                                float* __restrict__ scl,
                                float* __restrict__ shf) {
    int idx = blockIdx.x * blockDim.x + threadIdx.x;
    if (idx < 32) {
        float s = gamma[idx] * rsqrtf(rvar[idx] + 1e-5f);
        scl[idx] = s;
        shf[idx] = beta[idx] - rmean[idx] * s;
    }
    if (idx >= COUT * CIN) return;
    int c  = idx & 63;
    int oc = idx >> 6;
    int o  = oc & 31;      // BN channel
    int i  = oc >> 5;      // rotation index
    int r  = c >> 3, cc = c & 7;
    int sr = (r - i) & 7;              // jnp.roll along R axis
    int scch = sr * 8 + cc;            // source channel in original weight

    float acc[3][3] = {{0.f,0.f,0.f},{0.f,0.f,0.f},{0.f,0.f,0.f}};
    float th = (6.283185307179586f / 8.0f) * (float)i;
    float cs = cosf(th), sn = sinf(th);
    #pragma unroll
    for (int row = 0; row < 3; ++row) {
        #pragma unroll
        for (int col = 0; col < 3; ++col) {
            float fx = (float)(col - 1), fy = (float)(row - 1);
            int lx = (int)rintf(fx * cs + fy * sn);      // RNE matches np.around
            int ly = (int)rintf(-fx * sn + fy * cs);
            acc[1 - ly][1 - lx] += w[((o * CIN + scch) * 3 + row) * 3 + col];
        }
    }
    #pragma unroll
    for (int t = 0; t < 9; ++t)
        A[oc * KDIM + t * 64 + c] = (_Float16)acc[t / 3][t % 3];
}

// ---------------------------------------------------------------------------
// Phase B: implicit-GEMM conv (f16 WMMA, f32 accum) + fused BN + ReLU.
// One workgroup (8 waves) per (batch, output row): 256 oc x 96 positions.
// Fully unrolled k-loop: all DS/global loads use immediate offsets so the
// inner loop has no address VALU (avoids WMMA->VALU WAR hazard nops).
// ---------------------------------------------------------------------------
__global__ void __launch_bounds__(256)
rc_conv_kernel(const float* __restrict__ x,
               const _Float16* __restrict__ A,
               const float* __restrict__ scl,
               const float* __restrict__ shf,
               float* __restrict__ out) {
    __shared__ _Float16 lds[LDS_ELEMS];   // layout: [xi(98)][ky(3)][c padded to 72]

    int b = blockIdx.x / HH;
    int y = blockIdx.x % HH;
    int tid = threadIdx.x;

    // ---- stage 3 input rows (with halo) for all 64 channels, f32 -> f16 ----
    for (int idx = tid; idx < 3 * CIN * 98; idx += 256) {
        int xi  = idx % 98;
        int rc  = idx / 98;
        int c   = rc & 63;
        int row = rc >> 6;
        int gx = xi - 1, gy = y + row - 1;
        float v = 0.f;
        if ((unsigned)gx < (unsigned)WW && (unsigned)gy < (unsigned)HH)
            v = x[(((b * CIN) + c) * HH + gy) * WW + gx];
        lds[(xi * 3 + row) * ROWLEN + c] = (_Float16)v;
    }
    __syncthreads();

    int wv   = tid >> 5;
    int lane = tid & 31;
    int lrow = lane & 15;
    int hi   = lane >> 4;          // 0: lanes 0-15, 1: lanes 16-31
    int oc_base = wv * 32;

    v8f acc[2][6];
    v8f vzero = {0.f,0.f,0.f,0.f,0.f,0.f,0.f,0.f};
    #pragma unroll
    for (int m = 0; m < 2; ++m)
        #pragma unroll
        for (int j = 0; j < 6; ++j) acc[m][j] = vzero;

    // Per-lane bases; all per-step offsets below are compile-time constants.
    const _Float16* Ab0 = A + (size_t)(oc_base + lrow) * KDIM + hi * 8;
    const _Float16* Ab1 = A + (size_t)(oc_base + 16 + lrow) * KDIM + hi * 8;
    const _Float16* bbase = lds + lrow * (3 * ROWLEN) + hi * 8;

    #pragma unroll
    for (int s = 0; s < 18; ++s) {
        const int k0 = s * 32;
        const int t  = s >> 1;            // tap index (k = tap*64 + c)
        const int ky = t / 3, kx = t % 3;
        const int cc = (s & 1) * 32;      // channel sub-block within tap

        // A fragments (global, L2-resident weights), immediate offsets
        F16Frag a0, a1;
        a0.u[0] = *(const v4u*)(Ab0 + k0);
        a0.u[1] = *(const v4u*)(Ab0 + k0 + 16);
        a1.u[0] = *(const v4u*)(Ab1 + k0);
        a1.u[1] = *(const v4u*)(Ab1 + k0 + 16);

        // All 6 B fragments up front: ds_load_b128 with immediate offsets
        F16Frag bf[6];
        #pragma unroll
        for (int j = 0; j < 6; ++j) {
            const _Float16* lp = bbase + ((j * 16 + kx) * 3 + ky) * ROWLEN + cc;
            bf[j].u[0] = *(const v4u*)(lp);
            bf[j].u[1] = *(const v4u*)(lp + 16);
        }

        // 12 back-to-back WMMAs (distinct D, shared A/B: no data hazards)
        #pragma unroll
        for (int j = 0; j < 6; ++j) {
            acc[0][j] = __builtin_amdgcn_wmma_f32_16x16x32_f16(
                false, a0.h, false, bf[j].h, (short)0, acc[0][j], false, false);
            acc[1][j] = __builtin_amdgcn_wmma_f32_16x16x32_f16(
                false, a1.h, false, bf[j].h, (short)0, acc[1][j], false, false);
        }
    }

    // ---- epilogue: BN + ReLU + non-temporal store (output is write-once) ----
    float vs[2][8], vb[2][8];
    #pragma unroll
    for (int m = 0; m < 2; ++m)
        #pragma unroll
        for (int r = 0; r < 8; ++r) {
            int o = m * 16 + hi * 8 + r;     // oc_base is a multiple of 32 -> o == oc & 31
            vs[m][r] = scl[o];
            vb[m][r] = shf[o];
        }

    #pragma unroll
    for (int m = 0; m < 2; ++m)
        #pragma unroll
        for (int j = 0; j < 6; ++j) {
            int pos = j * 16 + lrow;
            #pragma unroll
            for (int r = 0; r < 8; ++r) {
                int ocg = oc_base + m * 16 + hi * 8 + r;
                float v = acc[m][j][r] * vs[m][r] + vb[m][r];
                v = fmaxf(v, 0.f);
                __builtin_nontemporal_store(
                    v, &out[(((size_t)b * COUT + ocg) * HH + y) * WW + pos]);
            }
        }
}

// ---------------------------------------------------------------------------
extern "C" void kernel_launch(void* const* d_in, const int* in_sizes, int n_in,
                              void* d_out, int out_size, void* d_ws, size_t ws_size,
                              hipStream_t stream) {
    (void)in_sizes; (void)n_in; (void)out_size; (void)ws_size;
    const float* x      = (const float*)d_in[0];
    const float* weight = (const float*)d_in[1];
    const float* gamma  = (const float*)d_in[2];
    const float* beta   = (const float*)d_in[3];
    const float* rmean  = (const float*)d_in[4];
    const float* rvar   = (const float*)d_in[5];
    float* out = (float*)d_out;

    _Float16* A  = (_Float16*)d_ws;                          // 256*576*2 = 294912 B
    float* scl   = (float*)((char*)d_ws + COUT * KDIM * 2);  // 32 floats
    float* shf   = scl + 32;

    rc_build_kernel<<<dim3(64), dim3(256), 0, stream>>>(weight, gamma, beta, rmean, rvar,
                                                        A, scl, shf);
    rc_conv_kernel<<<dim3(16 * HH), dim3(256), 0, stream>>>(x, A, scl, shf, out);
}
